// MixtralTransformerDecoder_84679575208321
// MI455X (gfx1250) — compile-verified, matrix operand order checked by reading, not run on hardware
//
#include <hip/hip_runtime.h>

// ---------------------------------------------------------------------------
// Types / bf16 helpers (bf16 kept as u16 storage; bit-cast into WMMA frags)
// ---------------------------------------------------------------------------
typedef __attribute__((ext_vector_type(16))) __bf16         v16bf;
typedef __attribute__((ext_vector_type(8)))  float          v8f;
typedef __attribute__((ext_vector_type(4)))  float          v4f;
typedef __attribute__((ext_vector_type(8)))  unsigned short v8us;
typedef __attribute__((ext_vector_type(4)))  unsigned int   v4u;
typedef __attribute__((ext_vector_type(8)))  int            v8i;
typedef __attribute__((ext_vector_type(4)))  int            v4i;

union Frag {
  v16bf bf;
  struct { v8us lo, hi; } u;
  unsigned short s[16];
};

__device__ __forceinline__ unsigned short f2bf(float f) {
  unsigned int u = __float_as_uint(f);
  unsigned int r = u + 0x7fffu + ((u >> 16) & 1u);   // round-to-nearest-even
  return (unsigned short)(r >> 16);
}
__device__ __forceinline__ float bf2f(unsigned short h) {
  return __uint_as_float(((unsigned int)h) << 16);
}
__device__ __forceinline__ v8f vzero8() { v8f z; for (int i = 0; i < 8; ++i) z[i] = 0.f; return z; }

#define T_TOK   4096      // B*S
#define HDIM    1024
#define SEQ     2048
#define NQH     16
#define NKVH    8
#define HD      64
#define IDIM    2048
#define NEXP    8
#define WIN     512
#define LDSS    40        // LDS row stride in u16: 80B -> 16B-aligned chunks, conflict-free

// ---------------------------------------------------------------------------
// TDM: issue a 128-row x 32-col bf16 tile load (global, row stride = rowlen
// elements) into LDS with padding so LDS rows are LDSS(=40) u16 apart.
// D# packing per CDNA5 ISA:
//   g0: count=1 | lds_addr | global_addr[56:0] | type=2
//   g1: data_size=1(2B), pad_enable, pad_interval=3(16 DW), pad_amount=3(4 DW),
//       tensor_dim0=rowlen, tensor_dim1=nrows, tile_dim0=32, tile_dim1=128,
//       tensor_dim0_stride=rowlen
//   g2/g3/g4: zero (2D tensor)
// Toolchain here exposes the 6-arg builtin: (v4u, v8i, v4i, v4i, v8i, i32 cpol)
// ---------------------------------------------------------------------------
__device__ __forceinline__ void tdm_load_tile_128x32(
    const unsigned short* gptr, void* lds_ptr, int rowlen, int nrows)
{
  unsigned long long ga = (unsigned long long)gptr;
  unsigned lds_off = (unsigned)(unsigned long long)lds_ptr;
  v4u g0;
  g0[0] = 1u;                                           // count=1 (valid user D#)
  g0[1] = lds_off;                                      // lds_addr
  g0[2] = (unsigned)ga;                                 // global_addr[31:0]
  g0[3] = (unsigned)((ga >> 32) & 0x01FFFFFFull) | (2u << 30);  // addr[56:32] | type=2
  v8i g1;
  g1[0] = (1 << 16) | (1 << 20) | (3 << 22) | (3 << 25);  // data_size|pad_en|pad_int|pad_amt
  unsigned d0 = (unsigned)rowlen, d1 = (unsigned)nrows;
  g1[1] = (int)((d0 & 0xFFFFu) << 16);                  // tensor_dim0[15:0]
  g1[2] = (int)(((d0 >> 16) & 0xFFFFu) | ((d1 & 0xFFFFu) << 16));
  g1[3] = (int)(((d1 >> 16) & 0xFFFFu) | (32u << 16));  // tile_dim0 = 32
  g1[4] = 128;                                          // tile_dim1 = 128
  g1[5] = rowlen;                                       // tensor_dim0_stride[31:0]
  g1[6] = 0;
  g1[7] = 0;
  v4i gz4; gz4[0] = 0; gz4[1] = 0; gz4[2] = 0; gz4[3] = 0;
  v8i gz8; for (int i = 0; i < 8; ++i) gz8[i] = 0;
  __builtin_amdgcn_tensor_load_to_lds(g0, g1, gz4, gz4, gz8, 0);
}

// ---------------------------------------------------------------------------
// RMSNorm: f32 [T,1024] -> bf16 [T,1024], one 256-thread block per row
// ---------------------------------------------------------------------------
__global__ __launch_bounds__(256) void rmsnorm_kernel(
    const float* __restrict__ x, const float* __restrict__ scale,
    unsigned short* __restrict__ out)
{
  __shared__ float red[256];
  const int row = blockIdx.x, tid = threadIdx.x;
  const float* xr = x + (size_t)row * HDIM;
  float vals[4], s = 0.f;
  #pragma unroll
  for (int i = 0; i < 4; ++i) { vals[i] = xr[tid + i * 256]; s += vals[i] * vals[i]; }
  red[tid] = s; __syncthreads();
  for (int off = 128; off > 0; off >>= 1) {
    if (tid < off) red[tid] += red[tid + off];
    __syncthreads();
  }
  const float inv = rsqrtf(red[0] * (1.f / HDIM) + 1e-5f);
  #pragma unroll
  for (int i = 0; i < 4; ++i)
    out[(size_t)row * HDIM + tid + i * 256] = f2bf(vals[i] * inv * scale[tid + i * 256]);
}

// ---------------------------------------------------------------------------
// Weight transpose+convert: in f32 [K,N] -> out bf16 [N,K] (K-contiguous so
// the GEMM can TDM-copy B tiles straight into LDS).
// ---------------------------------------------------------------------------
__global__ __launch_bounds__(256) void transpose_bf16_kernel(
    const float* __restrict__ in, unsigned short* __restrict__ out, int K, int N)
{
  __shared__ unsigned short t[32][33];
  const int k0 = blockIdx.y * 32, n0 = blockIdx.x * 32;
  const int tx = threadIdx.x & 31, ty = threadIdx.x >> 5;   // 32 x 8
  #pragma unroll
  for (int i = 0; i < 32; i += 8)
    t[ty + i][tx] = f2bf(in[(size_t)(k0 + ty + i) * N + n0 + tx]);
  __syncthreads();
  #pragma unroll
  for (int i = 0; i < 32; i += 8)
    out[(size_t)(n0 + ty + i) * K + k0 + tx] = t[tx][ty + i];
}

// ---------------------------------------------------------------------------
// bf16 WMMA GEMM: C[M,N] = A[M,K] * BT[N,K]^T, both bf16, both tiles staged
// into double-buffered LDS by the Tensor Data Mover (wave 0 issues, all waves
// sync on s_wait_tensorcnt + barrier). Block tile 128x128, K-step 32; 8 waves,
// each computes 32x64 via 8x v_wmma_f32_16x16x32_bf16 per k-step.
// MODE 0: store f32   1: store bf16   2: f32 +=   3: f32 += rowscale[row]*res
// MODE 4: bf16 store of silu(aux)*res   (fused SwiGLU epilogue)
// ---------------------------------------------------------------------------
template <int MODE>
__global__ __launch_bounds__(256) void gemm_wmma_kernel(
    const unsigned short* __restrict__ A, const unsigned short* __restrict__ BT,
    void* __restrict__ C, const float* __restrict__ aux,
    const float* __restrict__ rowscale, int M, int N, int K)
{
  __shared__ __attribute__((aligned(16))) unsigned short As[2][128 * LDSS];
  __shared__ __attribute__((aligned(16))) unsigned short Bs[2][128 * LDSS];
  const int tid  = threadIdx.x;
  const int lane = tid & 31, wid = tid >> 5;
  const int half = lane >> 4, l15 = lane & 15;
  const int wm = (wid & 3) * 32, wn = (wid >> 2) * 64;
  const int m0 = blockIdx.y * 128, n0 = blockIdx.x * 128;

  v8f acc[2][4];
  #pragma unroll
  for (int mi = 0; mi < 2; ++mi)
    #pragma unroll
    for (int ni = 0; ni < 4; ++ni) acc[mi][ni] = vzero8();

  const int nk = K >> 5;
  if (wid == 0) {
    tdm_load_tile_128x32(A  + (size_t)m0 * K, &As[0][0], K, M);
    tdm_load_tile_128x32(BT + (size_t)n0 * K, &Bs[0][0], K, N);
  }

  for (int ki = 0; ki < nk; ++ki) {
    const int cur = ki & 1;
    if (wid == 0) __builtin_amdgcn_s_wait_tensorcnt(0);
    __syncthreads();                      // LDS tile [cur] ready for everyone
    if (wid == 0 && ki + 1 < nk) {        // prefetch next k-step into [cur^1]
      const int k1 = (ki + 1) << 5;
      tdm_load_tile_128x32(A  + (size_t)m0 * K + k1, &As[cur ^ 1][0], K, M);
      tdm_load_tile_128x32(BT + (size_t)n0 * K + k1, &Bs[cur ^ 1][0], K, N);
    }
    Frag a[2], b[4];
    #pragma unroll
    for (int mi = 0; mi < 2; ++mi) {
      int row = wm + mi * 16 + l15;
      a[mi].u.lo = *(const v8us*)&As[cur][row * LDSS + half * 8];
      a[mi].u.hi = *(const v8us*)&As[cur][row * LDSS + 16 + half * 8];
    }
    #pragma unroll
    for (int ni = 0; ni < 4; ++ni) {
      int col = wn + ni * 16 + l15;
      b[ni].u.lo = *(const v8us*)&Bs[cur][col * LDSS + half * 8];
      b[ni].u.hi = *(const v8us*)&Bs[cur][col * LDSS + 16 + half * 8];
    }
    #pragma unroll
    for (int mi = 0; mi < 2; ++mi)
      #pragma unroll
      for (int ni = 0; ni < 4; ++ni)
        acc[mi][ni] = __builtin_amdgcn_wmma_f32_16x16x32_bf16(
            false, a[mi].bf, false, b[ni].bf, (short)0, acc[mi][ni], false, false);
    __syncthreads();                      // all reads of [cur] done before reuse
  }

  #pragma unroll
  for (int mi = 0; mi < 2; ++mi)
    #pragma unroll
    for (int ni = 0; ni < 4; ++ni)
      #pragma unroll
      for (int r = 0; r < 8; ++r) {
        int row = m0 + wm + mi * 16 + r + 8 * half;
        int col = n0 + wn + ni * 16 + l15;
        size_t o = (size_t)row * N + col;
        float res = acc[mi][ni][r];
        if constexpr (MODE == 0) ((float*)C)[o] = res;
        else if constexpr (MODE == 1) ((unsigned short*)C)[o] = f2bf(res);
        else if constexpr (MODE == 2) ((float*)C)[o] += res;
        else if constexpr (MODE == 3) ((float*)C)[o] += rowscale[row] * res;
        else {
          float g = aux[o];
          float si = g / (1.f + __expf(-g));
          ((unsigned short*)C)[o] = f2bf(si * res);
        }
      }
}

// ---------------------------------------------------------------------------
// RoPE (half-split convention), in-place on bf16 [T, nh, 64]
// ---------------------------------------------------------------------------
__global__ __launch_bounds__(256) void rope_kernel(unsigned short* __restrict__ x, int nh)
{
  int idx = blockIdx.x * 256 + threadIdx.x;
  int total = T_TOK * nh * 32;
  if (idx >= total) return;
  int d = idx & 31;
  int rem = idx >> 5;
  int h = rem % nh;
  int t = rem / nh;
  int pos = t & (SEQ - 1);
  float inv_freq = __powf(10000.f, -(float)d * (1.f / 32.f));
  float fr = (float)pos * inv_freq;
  float c = __cosf(fr), s = __sinf(fr);
  unsigned short* p = x + (size_t)t * nh * 64 + h * 64 + d;
  float x1 = bf2f(p[0]), x2 = bf2f(p[32]);
  p[0]  = f2bf(x1 * c - x2 * s);
  p[32] = f2bf(x2 * c + x1 * s);
}

// ---------------------------------------------------------------------------
// Sliding-window GQA flash attention. One wave per (b, head, 16-query tile).
// ---------------------------------------------------------------------------
__global__ __launch_bounds__(32) void attn_kernel(
    const unsigned short* __restrict__ q, const unsigned short* __restrict__ k,
    const unsigned short* __restrict__ v, unsigned short* __restrict__ out)
{
  __shared__ __attribute__((aligned(16))) unsigned short P[16 * LDSS];
  const int lane = threadIdx.x;
  const int half = lane >> 4, l15 = lane & 15;
  const int bid = blockIdx.x;
  const int qt = bid & 127;
  const int head = (bid >> 7) & 15;
  const int b = bid >> 11;
  const int kvh = head >> 1;            // NQ/NKV = 2
  const int q0 = qt * 16;

  Frag qa0, qa1;
  {
    const unsigned short* qr = q + ((size_t)(b * SEQ + q0 + l15)) * (NQH * HD) + head * HD;
    qa0.u.lo = *(const v8us*)&qr[half * 8];
    qa0.u.hi = *(const v8us*)&qr[16 + half * 8];
    qa1.u.lo = *(const v8us*)&qr[32 + half * 8];
    qa1.u.hi = *(const v8us*)&qr[48 + half * 8];
  }

  float m[8], l[8], al[8], rs[8];
  v8f o[4];
  #pragma unroll
  for (int r = 0; r < 8; ++r) { m[r] = -1e30f; l[r] = 0.f; }
  #pragma unroll
  for (int ni = 0; ni < 4; ++ni) o[ni] = vzero8();

  int jmin = q0 - (WIN - 1); if (jmin < 0) jmin = 0;
  const int kb0 = (jmin >> 5) << 5;

  for (int kb = kb0; kb <= q0 + 15; kb += 32) {
    v8f s0 = vzero8(), s1 = vzero8();
    {
      int key = kb + l15; if (key > SEQ - 1) key = SEQ - 1;
      const unsigned short* kr = k + ((size_t)(b * SEQ + key)) * (NKVH * HD) + kvh * HD;
      Frag kf0, kf1;
      kf0.u.lo = *(const v8us*)&kr[half * 8];
      kf0.u.hi = *(const v8us*)&kr[16 + half * 8];
      kf1.u.lo = *(const v8us*)&kr[32 + half * 8];
      kf1.u.hi = *(const v8us*)&kr[48 + half * 8];
      s0 = __builtin_amdgcn_wmma_f32_16x16x32_bf16(false, qa0.bf, false, kf0.bf, (short)0, s0, false, false);
      s0 = __builtin_amdgcn_wmma_f32_16x16x32_bf16(false, qa1.bf, false, kf1.bf, (short)0, s0, false, false);
    }
    {
      int key = kb + 16 + l15; if (key > SEQ - 1) key = SEQ - 1;
      const unsigned short* kr = k + ((size_t)(b * SEQ + key)) * (NKVH * HD) + kvh * HD;
      Frag kf0, kf1;
      kf0.u.lo = *(const v8us*)&kr[half * 8];
      kf0.u.hi = *(const v8us*)&kr[16 + half * 8];
      kf1.u.lo = *(const v8us*)&kr[32 + half * 8];
      kf1.u.hi = *(const v8us*)&kr[48 + half * 8];
      s1 = __builtin_amdgcn_wmma_f32_16x16x32_bf16(false, qa0.bf, false, kf0.bf, (short)0, s1, false, false);
      s1 = __builtin_amdgcn_wmma_f32_16x16x32_bf16(false, qa1.bf, false, kf1.bf, (short)0, s1, false, false);
    }
    #pragma unroll
    for (int r = 0; r < 8; ++r) {
      int i = q0 + r + 8 * half;
      int j0 = kb + l15, j1 = kb + 16 + l15;
      bool ok0 = (j0 <= i) && (i < j0 + WIN);
      bool ok1 = (j1 <= i) && (i < j1 + WIN);
      float v0 = ok0 ? s0[r] * 0.125f : -1e30f;   // 1/sqrt(64)
      float v1 = ok1 ? s1[r] * 0.125f : -1e30f;
      float mx = fmaxf(v0, v1);
      mx = fmaxf(mx, __shfl_xor(mx, 1));
      mx = fmaxf(mx, __shfl_xor(mx, 2));
      mx = fmaxf(mx, __shfl_xor(mx, 4));
      mx = fmaxf(mx, __shfl_xor(mx, 8));
      float mn = fmaxf(m[r], mx);
      al[r] = __expf(m[r] - mn);
      m[r] = mn;
      float p0 = ok0 ? __expf(v0 - mn) : 0.f;
      float p1 = ok1 ? __expf(v1 - mn) : 0.f;
      int prow = r + 8 * half;
      P[prow * LDSS + l15]      = f2bf(p0);
      P[prow * LDSS + 16 + l15] = f2bf(p1);
      float t = p0 + p1;
      t += __shfl_xor(t, 1); t += __shfl_xor(t, 2);
      t += __shfl_xor(t, 4); t += __shfl_xor(t, 8);
      rs[r] = t;
    }
    __syncthreads();
    Frag pa;
    pa.u.lo = *(const v8us*)&P[l15 * LDSS + half * 8];
    pa.u.hi = *(const v8us*)&P[l15 * LDSS + 16 + half * 8];
    #pragma unroll
    for (int ni = 0; ni < 4; ++ni) {
      Frag vf;
      int col = ni * 16 + l15;
      #pragma unroll
      for (int i = 0; i < 8; ++i) {
        int ka = kb + half * 8 + i;        if (ka > SEQ - 1) ka = SEQ - 1;
        int kc = kb + 16 + half * 8 + i;   if (kc > SEQ - 1) kc = SEQ - 1;
        vf.s[i]     = v[((size_t)(b * SEQ + ka)) * (NKVH * HD) + kvh * HD + col];
        vf.s[8 + i] = v[((size_t)(b * SEQ + kc)) * (NKVH * HD) + kvh * HD + col];
      }
      #pragma unroll
      for (int r = 0; r < 8; ++r) o[ni][r] *= al[r];
      o[ni] = __builtin_amdgcn_wmma_f32_16x16x32_bf16(false, pa.bf, false, vf.bf, (short)0, o[ni], false, false);
    }
    #pragma unroll
    for (int r = 0; r < 8; ++r) l[r] = l[r] * al[r] + rs[r];
    __syncthreads();
  }
  #pragma unroll
  for (int ni = 0; ni < 4; ++ni)
    #pragma unroll
    for (int r = 0; r < 8; ++r) {
      int row = q0 + r + 8 * half;
      out[((size_t)(b * SEQ + row)) * (NQH * HD) + head * HD + ni * 16 + l15] =
          f2bf(o[ni][r] / l[r]);
    }
}

// ---------------------------------------------------------------------------
// Router: logits = xn2[T,1024] @ w[1024,8]; softmax + top-2 + renorm.
// ---------------------------------------------------------------------------
__global__ __launch_bounds__(256) void router_kernel(
    const unsigned short* __restrict__ xn, const float* __restrict__ w,
    float* __restrict__ logits_out, float* __restrict__ routing)
{
  const int tid = threadIdx.x, lane = tid & 31, wid = tid >> 5;
  const int t = blockIdx.x * 8 + wid;
  const unsigned short* xr = xn + (size_t)t * HDIM;
  float acc[NEXP];
  #pragma unroll
  for (int e = 0; e < NEXP; ++e) acc[e] = 0.f;
  for (int h = lane; h < HDIM; h += 32) {
    float xv = bf2f(xr[h]);
    const float* wr = w + (size_t)h * NEXP;
    #pragma unroll
    for (int e = 0; e < NEXP; ++e) acc[e] += xv * wr[e];
  }
  #pragma unroll
  for (int e = 0; e < NEXP; ++e) {
    acc[e] += __shfl_xor(acc[e], 16); acc[e] += __shfl_xor(acc[e], 8);
    acc[e] += __shfl_xor(acc[e], 4);  acc[e] += __shfl_xor(acc[e], 2);
    acc[e] += __shfl_xor(acc[e], 1);
  }
  float mx = acc[0];
  #pragma unroll
  for (int e = 1; e < NEXP; ++e) mx = fmaxf(mx, acc[e]);
  float p[NEXP];
  #pragma unroll
  for (int e = 0; e < NEXP; ++e) p[e] = __expf(acc[e] - mx);
  int i1 = 0;
  #pragma unroll
  for (int e = 1; e < NEXP; ++e) if (p[e] > p[i1]) i1 = e;
  int i2 = (i1 == 0) ? 1 : 0;
  #pragma unroll
  for (int e = 0; e < NEXP; ++e) if (e != i1 && p[e] > p[i2]) i2 = e;
  float ws2 = p[i1] + p[i2];
  if (lane < NEXP) {
    logits_out[(size_t)t * NEXP + lane] = acc[lane];
    float r = (lane == i1) ? p[i1] / ws2 : (lane == i2) ? p[i2] / ws2 : 0.f;
    routing[(size_t)lane * T_TOK + t] = r;
  }
}

// ---------------------------------------------------------------------------
// Host launch
// ---------------------------------------------------------------------------
extern "C" void kernel_launch(void* const* d_in, const int* in_sizes, int n_in,
                              void* d_out, int out_size, void* d_ws, size_t ws_size,
                              hipStream_t stream)
{
  const float* x        = (const float*)d_in[0];
  const float* attn_ln  = (const float*)d_in[1];
  const float* wq       = (const float*)d_in[2];
  const float* wk       = (const float*)d_in[3];
  const float* wv       = (const float*)d_in[4];
  const float* wo       = (const float*)d_in[5];
  const float* ffn_ln   = (const float*)d_in[6];
  const float* router_w = (const float*)d_in[7];
  const float* w_gate   = (const float*)d_in[8];
  const float* w_up     = (const float*)d_in[9];
  const float* w_down   = (const float*)d_in[10];

  float* out        = (float*)d_out;                       // [T,1024]
  float* logits_out = out + (size_t)T_TOK * HDIM;          // [T,8]

  char* ws = (char*)d_ws;
  size_t off = 0;
  auto alloc = [&](size_t bytes) { void* p = ws + off; off += (bytes + 255) & ~(size_t)255; return p; };
  unsigned short* xn   = (unsigned short*)alloc((size_t)T_TOK * HDIM * 2);
  unsigned short* qb   = (unsigned short*)alloc((size_t)T_TOK * NQH * HD * 2);
  unsigned short* kb   = (unsigned short*)alloc((size_t)T_TOK * NKVH * HD * 2);
  unsigned short* vb   = (unsigned short*)alloc((size_t)T_TOK * NKVH * HD * 2);
  unsigned short* atb  = (unsigned short*)alloc((size_t)T_TOK * HDIM * 2);
  float*          xbuf = (float*)alloc((size_t)T_TOK * HDIM * 4);
  unsigned short* xn2  = (unsigned short*)alloc((size_t)T_TOK * HDIM * 2);
  float*          rout = (float*)alloc((size_t)NEXP * T_TOK * 4);
  float*          gbuf = (float*)alloc((size_t)T_TOK * IDIM * 4);
  unsigned short* actb = (unsigned short*)alloc((size_t)T_TOK * IDIM * 2);
  unsigned short* wT   = (unsigned short*)alloc((size_t)IDIM * HDIM * 2);   // 4MB, reused

  const dim3 blk256(256);
  // ---- attention block ----
  rmsnorm_kernel<<<T_TOK, blk256, 0, stream>>>(x, attn_ln, xn);
  transpose_bf16_kernel<<<dim3(32, 32), blk256, 0, stream>>>(wq, wT, 1024, 1024);
  gemm_wmma_kernel<1><<<dim3(8, 32),  blk256, 0, stream>>>(xn, wT, qb, nullptr, nullptr, T_TOK, 1024, 1024);
  transpose_bf16_kernel<<<dim3(16, 32), blk256, 0, stream>>>(wk, wT, 1024, 512);
  gemm_wmma_kernel<1><<<dim3(4, 32),  blk256, 0, stream>>>(xn, wT, kb, nullptr, nullptr, T_TOK, 512, 1024);
  transpose_bf16_kernel<<<dim3(16, 32), blk256, 0, stream>>>(wv, wT, 1024, 512);
  gemm_wmma_kernel<1><<<dim3(4, 32),  blk256, 0, stream>>>(xn, wT, vb, nullptr, nullptr, T_TOK, 512, 1024);
  rope_kernel<<<(T_TOK * NQH * 32 + 255) / 256,  blk256, 0, stream>>>(qb, NQH);
  rope_kernel<<<(T_TOK * NKVH * 32 + 255) / 256, blk256, 0, stream>>>(kb, NKVH);
  attn_kernel<<<2 * NQH * (SEQ / 16), dim3(32), 0, stream>>>(qb, kb, vb, atb);
  (void)hipMemcpyAsync(xbuf, x, (size_t)T_TOK * HDIM * 4, hipMemcpyDeviceToDevice, stream);
  transpose_bf16_kernel<<<dim3(32, 32), blk256, 0, stream>>>(wo, wT, 1024, 1024);
  gemm_wmma_kernel<2><<<dim3(8, 32), blk256, 0, stream>>>(atb, wT, xbuf, nullptr, nullptr, T_TOK, 1024, 1024);

  // ---- MoE block ----
  rmsnorm_kernel<<<T_TOK, blk256, 0, stream>>>(xbuf, ffn_ln, xn2);
  router_kernel<<<T_TOK / 8, blk256, 0, stream>>>(xn2, router_w, logits_out, rout);
  (void)hipMemcpyAsync(out, xbuf, (size_t)T_TOK * HDIM * 4, hipMemcpyDeviceToDevice, stream);
  for (int e = 0; e < NEXP; ++e) {
    const float* wg = w_gate + (size_t)e * HDIM * IDIM;
    const float* wu = w_up   + (size_t)e * HDIM * IDIM;
    const float* wd = w_down + (size_t)e * IDIM * HDIM;
    transpose_bf16_kernel<<<dim3(64, 32), blk256, 0, stream>>>(wg, wT, 1024, 2048);
    gemm_wmma_kernel<0><<<dim3(16, 32), blk256, 0, stream>>>(xn2, wT, gbuf, nullptr, nullptr, T_TOK, IDIM, HDIM);
    transpose_bf16_kernel<<<dim3(64, 32), blk256, 0, stream>>>(wu, wT, 1024, 2048);
    gemm_wmma_kernel<4><<<dim3(16, 32), blk256, 0, stream>>>(xn2, wT, actb, gbuf, nullptr, T_TOK, IDIM, HDIM);
    transpose_bf16_kernel<<<dim3(32, 64), blk256, 0, stream>>>(wd, wT, 2048, 1024);
    gemm_wmma_kernel<3><<<dim3(8, 32),  blk256, 0, stream>>>(actb, wT, out, nullptr, rout + (size_t)e * T_TOK, T_TOK, HDIM, IDIM);
  }
}
